// HierarchicalJointModelGloVeBERT_51685636440074
// MI455X (gfx1250) — matrix operand order repaired and checked by reading, not compiled, not run on hardware
//
#include <hip/hip_runtime.h>

// ---------------- types ----------------
typedef unsigned short bf16_t;
typedef __attribute__((ext_vector_type(16))) __bf16 v16bf;
typedef __attribute__((ext_vector_type(8)))  float  v8f;

#define HDIM 256
#define GDIM 768   // 3*H

__device__ __forceinline__ bf16_t f2bf(float f) {
  unsigned int u = __float_as_uint(f);
  u += 0x7FFFu + ((u >> 16) & 1u);   // round-to-nearest-even
  return (bf16_t)(u >> 16);
}

union FragB { uint4 u[2]; v16bf v; };

// ---------------- WMMA GEMM ----------------
// C[M,N] = A(bf16 rows via aStride/aOff) * Bpack + bias, f32 out.
// Bpack layout: tiles [nt][kt][lane 0..31][elem 0..15] bf16 where
//   lane<16 : B[k=kt*32+(e>>3)*16+(e&7),   n=nt*16+lane]
//   lane>=16: B[k=kt*32+(e>>3)*16+8+(e&7), n=nt*16+lane-16]
// Each wave computes a 16(M) x 64(N) tile (4 accumulators, A reuse).
// Latency is hidden via wave-level parallelism (~72 VGPRs -> high occupancy);
// the compiler clause-groups the b128 loads and unrolls the K loop.
__global__ void k_gemm_bf16(const bf16_t* __restrict__ A, long aStride, long aOff,
                            const bf16_t* __restrict__ Bp, const float* __restrict__ bias,
                            float* __restrict__ C, int ldc, int M, int KT) {
  const int lane  = threadIdx.x;
  const int tileM = blockIdx.y * blockDim.y + threadIdx.y;
  if (tileM * 16 >= M) return;
  const int tileN0 = blockIdx.x * 4;

  const int mrow = tileM * 16 + (lane & 15);
  const int klo  = (lane >> 4) * 8;           // A: lanes 0-15 K{0..7,16..23}, 16-31 K{8..15,24..31}
  const bf16_t* arow = A + (long)mrow * aStride + aOff;

  v8f acc[4];
  #pragma unroll
  for (int i = 0; i < 4; ++i)
    #pragma unroll
    for (int r = 0; r < 8; ++r) acc[i][r] = 0.f;

  for (int kt = 0; kt < KT; ++kt) {
    FragB a;
    a.u[0] = *(const uint4*)(arow + (long)kt * 32 + klo);
    a.u[1] = *(const uint4*)(arow + (long)kt * 32 + 16 + klo);
    #pragma unroll
    for (int i = 0; i < 4; ++i) {
      FragB b;
      const uint4* bq = (const uint4*)(Bp + (((long)(tileN0 + i) * KT + kt) * 32 + lane) * 16);
      b.u[0] = bq[0];
      b.u[1] = bq[1];
      acc[i] = __builtin_amdgcn_wmma_f32_16x16x32_bf16(
          false, a.v, false, b.v, (short)0, acc[i], false, false);
    }
  }

  // D layout: elem r of lane L: L<16 -> (M=r, N=L); L>=16 -> (M=8+r, N=L-16)
  const int nlo   = lane & 15;
  const int mbase = tileM * 16 + ((lane >> 4) * 8);
  #pragma unroll
  for (int i = 0; i < 4; ++i) {
    int ncol = (tileN0 + i) * 16 + nlo;
    float bv = bias ? bias[ncol] : 0.f;
    float* cp = C + (long)mbase * ldc + ncol;
    #pragma unroll
    for (int r = 0; r < 8; ++r) cp[(long)r * ldc] = acc[i][r] + bv;
  }
}

// ---------------- weight pack: W[N,Kreal] row-major -> WMMA B fragments ----------------
__global__ void k_pack_b(const float* __restrict__ W, int Kreal, int KT,
                         bf16_t* __restrict__ out, long total) {
  long i = (long)blockIdx.x * blockDim.x + threadIdx.x;
  if (i >= total) return;
  int e    = (int)(i & 15);
  int lane = (int)((i >> 4) & 31);
  long tile = i >> 9;                 // nt*KT + kt
  int kt = (int)(tile % KT);
  int nt = (int)(tile / KT);
  int k  = kt * 32 + (e >> 3) * 16 + (lane >> 4) * 8 + (e & 7);
  int n  = nt * 16 + (lane & 15);
  float v = (k < Kreal) ? W[(long)n * Kreal + k] : 0.f;
  out[i] = f2bf(v);
}

// ---------------- embedding gather + bf16 cast (K padded 300->320) ----------------
__global__ void k_embed(const int* __restrict__ ids, const float* __restrict__ emb,
                        bf16_t* __restrict__ atok) {
  long tok = blockIdx.x;
  int  k   = threadIdx.x;              // 320 threads
  int  id  = ids[tok];
  float v  = (k < 300) ? emb[(long)id * 300 + k] : 0.f;
  atok[tok * 320 + k] = f2bf(v);
}

// ---------------- masked GRU gate update ----------------
__global__ void k_gru_gate(const float* __restrict__ xp, const float* __restrict__ hp,
                           float* __restrict__ h, bf16_t* __restrict__ hb,
                           const int* __restrict__ mask, int mstride, int moff, int M) {
  long i = (long)blockIdx.x * blockDim.x + threadIdx.x;
  long total = (long)M * HDIM;
  if (i >= total) return;
  long m = i >> 8;
  int  j = (int)(i & 255);
  const float* xr = xp + m * GDIM;
  const float* hr = hp + m * GDIM;
  float r = 1.f / (1.f + __expf(-(xr[j]          + hr[j])));
  float z = 1.f / (1.f + __expf(-(xr[HDIM + j]   + hr[HDIM + j])));
  float n = tanhf(xr[2 * HDIM + j] + r * hr[2 * HDIM + j]);
  float hold = h[i];
  float hnew = (1.f - z) * n + z * hold;
  bool valid = mask ? (mask[m * mstride + moff] != 0) : true;
  float ho = valid ? hnew : hold;
  h[i]  = ho;
  hb[i] = f2bf(ho);
}

// ---------------- small utilities ----------------
__global__ void k_zero_f32(float* p, long n) {
  long i = (long)blockIdx.x * blockDim.x + threadIdx.x;
  if (i < n) p[i] = 0.f;
}
__global__ void k_zero_b16(bf16_t* p, long n) {
  long i = (long)blockIdx.x * blockDim.x + threadIdx.x;
  if (i < n) p[i] = 0;
}
__global__ void k_copy_h_bf16(const float* __restrict__ h, bf16_t* __restrict__ dst,
                              int ld, int colOff, long total) {
  long i = (long)blockIdx.x * blockDim.x + threadIdx.x;
  if (i >= total) return;
  long m = i >> 8; int j = (int)(i & 255);
  dst[m * ld + colOff + j] = f2bf(h[i]);
}
__global__ void k_copy_h_f32(const float* __restrict__ h, float* __restrict__ dst,
                             int ld, int colOff, long total) {
  long i = (long)blockIdx.x * blockDim.x + threadIdx.x;
  if (i >= total) return;
  long m = i >> 8; int j = (int)(i & 255);
  dst[m * ld + colOff + j] = h[i];
}
__global__ void k_rmask(const int* __restrict__ ids, int* __restrict__ rmask) {
  int m = blockIdx.x * blockDim.x + threadIdx.x;
  if (m >= 8192) return;
  const int* p = ids + (long)m * 32;
  int a = 0;
  for (int w = 0; w < 32; ++w) a |= p[w];
  rmask[m] = a;
}
__global__ void k_pbt_fill(const float* __restrict__ bert, const float* __restrict__ uf,
                           const float* __restrict__ ufw, float* __restrict__ pbt) {
  long i = (long)blockIdx.x * blockDim.x + threadIdx.x;
  if (i >= 512L * 788) return;
  int m = (int)(i / 788);
  int c = (int)(i % 788);
  if (c < 768) pbt[(long)m * 1300 + 512 + c] = bert[(long)m * 768 + c];
  else { int u = c - 768; pbt[(long)m * 1300 + 1280 + u] = uf[(long)m * 20 + u] * ufw[u]; }
}
// LayerNorm (unbiased std), writes bf16 padded to 1312 cols
__global__ void k_layernorm(const float* __restrict__ x, const float* __restrict__ g,
                            const float* __restrict__ b, bf16_t* __restrict__ yb) {
  int m = blockIdx.x, tid = threadIdx.x;
  __shared__ float s1[256], s2[256];
  float s = 0.f, q = 0.f;
  for (int c = tid; c < 1300; c += 256) { float v = x[(long)m * 1300 + c]; s += v; q += v * v; }
  s1[tid] = s; s2[tid] = q; __syncthreads();
  for (int w = 128; w > 0; w >>= 1) {
    if (tid < w) { s1[tid] += s1[tid + w]; s2[tid] += s2[tid + w]; }
    __syncthreads();
  }
  float mu  = s1[0] / 1300.f;
  float var = (s2[0] - 1300.f * mu * mu) / 1299.f;
  float inv = 1.f / (sqrtf(fmaxf(var, 0.f)) + 1e-6f);
  for (int c = tid; c < 1312; c += 256) {
    if (c < 1300) yb[(long)m * 1312 + c] = f2bf(g[c] * (x[(long)m * 1300 + c] - mu) * inv + b[c]);
    else          yb[(long)m * 1312 + c] = 0;
  }
}
__global__ void k_rstars(const float* __restrict__ racc, const float* __restrict__ w2,
                         const float* __restrict__ b2, float* __restrict__ out) {
  int m = blockIdx.x, j = threadIdx.x;          // 512 blocks x 128 threads
  float v = racc[(long)m * 128 + j];
  v = v > 0.f ? v : expm1f(v);
  __shared__ float sm[128];
  sm[j] = v * w2[j]; __syncthreads();
  for (int w = 64; w > 0; w >>= 1) { if (j < w) sm[j] += sm[j + w]; __syncthreads(); }
  if (j == 0) out[m] = sm[0] + b2[0];
}
__global__ void k_pstars(const float* __restrict__ ph, const float* __restrict__ w1,
                         const float* __restrict__ b1, const float* __restrict__ w2,
                         const float* __restrict__ b2, float* __restrict__ out) {
  int p = blockIdx.x, j = threadIdx.x;          // 32 blocks x 128 threads
  float a = b1[j];
  const float* hv = ph + (long)p * 512;
  const float* wr = w1 + (long)j * 512;
  for (int k = 0; k < 512; ++k) a += hv[k] * wr[k];
  a = a > 0.f ? a : expm1f(a);
  __shared__ float sm[128];
  sm[j] = a; __syncthreads();
  if (j < 9) {
    float o = b2[j];
    for (int k = 0; k < 128; ++k) o += sm[k] * w2[j * 128 + k];
    out[(long)p * 9 + j] = o;
  }
}

// ---------------- host helpers ----------------
static inline void launch_gemm(hipStream_t s, const bf16_t* A, long aStride, long aOff,
                               const bf16_t* Bp, const float* bias, float* C, int ldc,
                               int M, int N, int KT) {
  dim3 grid(N / 64, (M / 16 + 3) / 4);
  dim3 block(32, 4);
  k_gemm_bf16<<<grid, block, 0, s>>>(A, aStride, aOff, Bp, bias, C, ldc, M, KT);
}
static inline void launch_pack(hipStream_t s, const float* W, int N, int Kreal, int KT, bf16_t* out) {
  long total = (long)(N / 16) * KT * 512;
  k_pack_b<<<(unsigned)((total + 255) / 256), 256, 0, s>>>(W, Kreal, KT, out, total);
}
// one GRU direction: per-step GEMM(x_t*Wih^T) + GEMM(h*Whh^T) + gate update
static void run_gru_dir(hipStream_t s, const bf16_t* A, int Kld, int KTx, int T, int M, bool rev,
                        const bf16_t* Wih, const float* bih, const bf16_t* Whh, const float* bhh,
                        const int* mask, int mstride,
                        float* xacc, float* hacc, float* h, bf16_t* hb) {
  long hn = (long)M * HDIM;
  unsigned hb_blocks = (unsigned)((hn + 255) / 256);
  k_zero_f32<<<hb_blocks, 256, 0, s>>>(h, hn);
  k_zero_b16<<<hb_blocks, 256, 0, s>>>(hb, hn);
  for (int i = 0; i < T; ++i) {
    int t = rev ? (T - 1 - i) : i;
    launch_gemm(s, A, (long)T * Kld, (long)t * Kld, Wih, bih, xacc, GDIM, M, GDIM, KTx);
    launch_gemm(s, hb, HDIM, 0, Whh, bhh, hacc, GDIM, M, GDIM, HDIM / 32);
    k_gru_gate<<<hb_blocks, 256, 0, s>>>(xacc, hacc, h, hb, mask, mstride, t, M);
  }
}

extern "C" void kernel_launch(void* const* d_in, const int* in_sizes, int n_in,
                              void* d_out, int out_size, void* d_ws, size_t ws_size,
                              hipStream_t stream) {
  (void)in_sizes; (void)n_in; (void)out_size; (void)ws_size;
  const int*   ids   = (const int*)  d_in[0];
  const float* bert  = (const float*)d_in[1];
  const float* ufeat = (const float*)d_in[2];
  const float* emb   = (const float*)d_in[3];
  const float *s_wih_f=(const float*)d_in[4],  *s_whh_f=(const float*)d_in[5],
              *s_bih_f=(const float*)d_in[6],  *s_bhh_f=(const float*)d_in[7];
  const float *s_wih_b=(const float*)d_in[8],  *s_whh_b=(const float*)d_in[9],
              *s_bih_b=(const float*)d_in[10], *s_bhh_b=(const float*)d_in[11];
  const float *r_wih_f=(const float*)d_in[12], *r_whh_f=(const float*)d_in[13],
              *r_bih_f=(const float*)d_in[14], *r_bhh_f=(const float*)d_in[15];
  const float *r_wih_b=(const float*)d_in[16], *r_whh_b=(const float*)d_in[17],
              *r_bih_b=(const float*)d_in[18], *r_bhh_b=(const float*)d_in[19];
  const float *p_wih_f=(const float*)d_in[20], *p_whh_f=(const float*)d_in[21],
              *p_bih_f=(const float*)d_in[22], *p_bhh_f=(const float*)d_in[23];
  const float *p_wih_b=(const float*)d_in[24], *p_whh_b=(const float*)d_in[25],
              *p_bih_b=(const float*)d_in[26], *p_bhh_b=(const float*)d_in[27];
  const float *ln_g=(const float*)d_in[28], *ln_b=(const float*)d_in[29];
  const float *rff_w1=(const float*)d_in[30], *rff_b1=(const float*)d_in[31],
              *rff_w2=(const float*)d_in[32], *rff_b2=(const float*)d_in[33];
  const float *pff_w1=(const float*)d_in[34], *pff_b1=(const float*)d_in[35],
              *pff_w2=(const float*)d_in[36], *pff_b2=(const float*)d_in[37];
  const float *ufw = (const float*)d_in[38];
  float* out = (float*)d_out;

  // workspace carve-up
  char* base = (char*)d_ws;
  size_t off = 0;
  auto alloc = [&](size_t bytes) -> void* {
    void* p = base + off;
    off = (off + bytes + 255) & ~(size_t)255;
    return p;
  };
  bf16_t* ATOK = (bf16_t*)alloc(262144L * 320 * 2);   // embedded tokens, K padded
  float*  XACC = (float*) alloc(8192L * GDIM * 4);    // x-projection accumulator
  float*  HACC = (float*) alloc(8192L * GDIM * 4);    // h-projection accumulator
  float*  Hbuf = (float*) alloc(8192L * HDIM * 4);
  bf16_t* HB   = (bf16_t*)alloc(8192L * HDIM * 2);
  bf16_t* SENT = (bf16_t*)alloc(8192L * 512 * 2);     // sentence BiGRU outputs
  int*    RMSK = (int*)   alloc(8192L * 4);
  float*  PBT  = (float*) alloc(512L * 1300 * 4);     // pre-LN concat
  bf16_t* PBTB = (bf16_t*)alloc(512L * 1312 * 2);     // post-LN bf16, K padded
  float*  PH   = (float*) alloc(32L * 512 * 4);       // product BiGRU final
  float*  RACC = (float*) alloc(512L * 128 * 4);
  bf16_t* SWF = (bf16_t*)alloc(48L * 10 * 512 * 2);
  bf16_t* SWB = (bf16_t*)alloc(48L * 10 * 512 * 2);
  bf16_t* SUF = (bf16_t*)alloc(48L * 8  * 512 * 2);
  bf16_t* SUB = (bf16_t*)alloc(48L * 8  * 512 * 2);
  bf16_t* RWF = (bf16_t*)alloc(48L * 16 * 512 * 2);
  bf16_t* RWB = (bf16_t*)alloc(48L * 16 * 512 * 2);
  bf16_t* RUF = (bf16_t*)alloc(48L * 8  * 512 * 2);
  bf16_t* RUB = (bf16_t*)alloc(48L * 8  * 512 * 2);
  bf16_t* PWF = (bf16_t*)alloc(48L * 41 * 512 * 2);
  bf16_t* PWB = (bf16_t*)alloc(48L * 41 * 512 * 2);
  bf16_t* PUF = (bf16_t*)alloc(48L * 8  * 512 * 2);
  bf16_t* PUB = (bf16_t*)alloc(48L * 8  * 512 * 2);
  bf16_t* QW  = (bf16_t*)alloc(8L  * 41 * 512 * 2);

  // pack all weights into WMMA B-fragment layout
  launch_pack(stream, s_wih_f, GDIM, 300, 10, SWF);
  launch_pack(stream, s_wih_b, GDIM, 300, 10, SWB);
  launch_pack(stream, s_whh_f, GDIM, 256, 8,  SUF);
  launch_pack(stream, s_whh_b, GDIM, 256, 8,  SUB);
  launch_pack(stream, r_wih_f, GDIM, 512, 16, RWF);
  launch_pack(stream, r_wih_b, GDIM, 512, 16, RWB);
  launch_pack(stream, r_whh_f, GDIM, 256, 8,  RUF);
  launch_pack(stream, r_whh_b, GDIM, 256, 8,  RUB);
  launch_pack(stream, p_wih_f, GDIM, 1300, 41, PWF);
  launch_pack(stream, p_wih_b, GDIM, 1300, 41, PWB);
  launch_pack(stream, p_whh_f, GDIM, 256, 8,  PUF);
  launch_pack(stream, p_whh_b, GDIM, 256, 8,  PUB);
  launch_pack(stream, rff_w1, 128, 1300, 41, QW);

  // embedding gather + masks
  k_embed<<<262144, 320, 0, stream>>>(ids, emb, ATOK);
  k_rmask<<<32, 256, 0, stream>>>(ids, RMSK);

  // ---- sentence level BiGRU: 8192 seqs, T=32, E=300 ----
  run_gru_dir(stream, ATOK, 320, 10, 32, 8192, false, SWF, s_bih_f, SUF, s_bhh_f,
              ids, 32, XACC, HACC, Hbuf, HB);
  k_copy_h_bf16<<<(8192 * 256 + 255) / 256, 256, 0, stream>>>(Hbuf, SENT, 512, 0, 8192L * 256);
  run_gru_dir(stream, ATOK, 320, 10, 32, 8192, true, SWB, s_bih_b, SUB, s_bhh_b,
              ids, 32, XACC, HACC, Hbuf, HB);
  k_copy_h_bf16<<<(8192 * 256 + 255) / 256, 256, 0, stream>>>(Hbuf, SENT, 512, 256, 8192L * 256);

  // ---- review level BiGRU: 512 seqs, T=16, D=512 ----
  run_gru_dir(stream, SENT, 512, 16, 16, 512, false, RWF, r_bih_f, RUF, r_bhh_f,
              RMSK, 16, XACC, HACC, Hbuf, HB);
  k_copy_h_f32<<<(512 * 256 + 255) / 256, 256, 0, stream>>>(Hbuf, PBT, 1300, 0, 512L * 256);
  run_gru_dir(stream, SENT, 512, 16, 16, 512, true, RWB, r_bih_b, RUB, r_bhh_b,
              RMSK, 16, XACC, HACC, Hbuf, HB);
  k_copy_h_f32<<<(512 * 256 + 255) / 256, 256, 0, stream>>>(Hbuf, PBT, 1300, 256, 512L * 256);

  // ---- concat + layer norm ----
  k_pbt_fill<<<(512 * 788 + 255) / 256, 256, 0, stream>>>(bert, ufeat, ufw, PBT);
  k_layernorm<<<512, 256, 0, stream>>>(PBT, ln_g, ln_b, PBTB);

  // ---- review FF head -> r_stars at out[288..799] ----
  launch_gemm(stream, PBTB, 1312, 0, QW, rff_b1, RACC, 128, 512, 128, 41);
  k_rstars<<<512, 128, 0, stream>>>(RACC, rff_w2, rff_b2, out + 288);

  // ---- product level BiGRU: 32 seqs, T=16, D=1300 (all-ones mask) ----
  run_gru_dir(stream, PBTB, 1312, 41, 16, 32, false, PWF, p_bih_f, PUF, p_bhh_f,
              nullptr, 0, XACC, HACC, Hbuf, HB);
  k_copy_h_f32<<<(32 * 256 + 255) / 256, 256, 0, stream>>>(Hbuf, PH, 512, 0, 32L * 256);
  run_gru_dir(stream, PBTB, 1312, 41, 16, 32, true, PWB, p_bih_b, PUB, p_bhh_b,
              nullptr, 0, XACC, HACC, Hbuf, HB);
  k_copy_h_f32<<<(32 * 256 + 255) / 256, 256, 0, stream>>>(Hbuf, PH, 512, 256, 32L * 256);

  // ---- product FF head -> p_stars at out[0..287] ----
  k_pstars<<<32, 128, 0, stream>>>(PH, pff_w1, pff_b1, pff_w2, pff_b2, out);
}
